// RWKV6Attention_85461259255809
// MI455X (gfx1250) — compile-verified, matrix operand order checked by reading, not run on hardware
//
#include <hip/hip_runtime.h>
#include <hip/hip_bf16.h>
#include <cstdint>
#include <cstddef>

// Problem constants (from reference)
#define B_   4
#define T_   2048
#define D_   1024
#define H_   16
#define DK_  64
#define DV_  128
#define K_   1024          // H*DK
#define V_   2048          // H*DV
#define R_   32
#define BT_  (B_*T_)
#define EPS_ 1e-5f

typedef __bf16 bf16;
typedef __attribute__((ext_vector_type(16))) __bf16 v16bf;
typedef __attribute__((ext_vector_type(8)))  float  v8f;

union FragU { v16bf v; uint4 q[2]; };

__device__ __forceinline__ v16bf load_frag(const bf16* base) {
    FragU f;
    f.q[0] = *reinterpret_cast<const uint4*>(base);
    f.q[1] = *reinterpret_cast<const uint4*>(base + 16);
    return f.v;
}

// CDNA5 async global->LDS copy, 16 bytes, tracked by ASYNCcnt.
__device__ __forceinline__ void async_copy_b128(uint32_t lds_off, const bf16* gsrc) {
    asm volatile("global_load_async_to_lds_b128 %0, %1, off"
                 :: "v"(lds_off), "v"((unsigned long long)(uintptr_t)gsrc)
                 : "memory");
}
__device__ __forceinline__ void wait_async0() {
    asm volatile("s_wait_asynccnt 0x0" ::: "memory");
}

template<int EPI>
__device__ __forceinline__ void epilogue_store(void* Out, const float* bias,
                                               int N, int mb, int n, const v8f& a) {
    float bval = 0.f;
    if (EPI == 2 || EPI == 3) bval = bias[n];
    #pragma unroll
    for (int e = 0; e < 8; ++e) {
        const float val = a[e];
        const size_t idx = (size_t)(mb + e) * N + n;
        if      (EPI == 0) ((bf16*)Out)[idx]  = (bf16)val;
        else if (EPI == 1) ((bf16*)Out)[idx]  = (bf16)tanhf(val);
        else if (EPI == 2) ((float*)Out)[idx] = val + bval;
        else if (EPI == 3) ((float*)Out)[idx] = __expf(-__expf(val + bval));
        else               ((float*)Out)[idx] = val;
    }
}

// ---------------------------------------------------------------------------
// LDS-staged bf16 WMMA GEMM: C(M,N) = A(M,Kd) * Bt(N,Kd)^T  (Bt col-major B)
// Block = 256 threads = 8 waves; block tile 128x64; wave tile 32x32.
// Double-buffered async global->LDS staging (ASYNCcnt), ds_load fragments.
// Requires M%128==0, N%64==0, Kd%32==0.
// ---------------------------------------------------------------------------
#define BM 128
#define BN 64
#define BK 32
#define LDS_STRIDE 40                  // 32 data + 8 pad elements (80B rows)
#define A_ELEMS (BM * LDS_STRIDE)      // 5120
#define B_ELEMS (BN * LDS_STRIDE)      // 2560

template<int EPI>
__global__ void __launch_bounds__(256)
gemm_bf16_wmma_lds(const bf16* __restrict__ A, const bf16* __restrict__ Bt,
                   void* __restrict__ Out, const float* __restrict__ bias,
                   int M, int N, int Kd) {
    __shared__ bf16 smem[2][A_ELEMS + B_ELEMS];

    const int tid  = threadIdx.x;
    const int lane = tid & 31;
    const int wave = tid >> 5;
    const int l15  = lane & 15;
    const int hlf  = lane >> 4;

    const int nblk = N / BN;
    const int bm = (blockIdx.x / nblk) * BM;
    const int bn = (blockIdx.x % nblk) * BN;

    const int wm = (wave >> 1) * 32;   // 0,32,64,96
    const int wn = (wave & 1) * 32;    // 0,32

    // Staging assignment: A tile = 512 x 16B chunks (2/thread), B = 256 (1/thread)
    const int ca0 = tid * 2, ca1 = tid * 2 + 1;
    const int ra0 = ca0 >> 2, qa0 = ca0 & 3;
    const int ra1 = ca1 >> 2, qa1 = ca1 & 3;
    const int rb0 = tid >> 2, qb0 = tid & 3;

    const bf16* gA0 = A  + (size_t)(bm + ra0) * Kd + qa0 * 8;
    const bf16* gA1 = A  + (size_t)(bm + ra1) * Kd + qa1 * 8;
    const bf16* gB0 = Bt + (size_t)(bn + rb0) * Kd + qb0 * 8;

    const uint32_t la0[2] = { (uint32_t)(uintptr_t)&smem[0][ra0 * LDS_STRIDE + qa0 * 8],
                              (uint32_t)(uintptr_t)&smem[1][ra0 * LDS_STRIDE + qa0 * 8] };
    const uint32_t la1[2] = { (uint32_t)(uintptr_t)&smem[0][ra1 * LDS_STRIDE + qa1 * 8],
                              (uint32_t)(uintptr_t)&smem[1][ra1 * LDS_STRIDE + qa1 * 8] };
    const uint32_t lb0[2] = { (uint32_t)(uintptr_t)&smem[0][A_ELEMS + rb0 * LDS_STRIDE + qb0 * 8],
                              (uint32_t)(uintptr_t)&smem[1][A_ELEMS + rb0 * LDS_STRIDE + qb0 * 8] };

    v8f acc[2][2] = {{}, {}};

    // prologue stage into buffer 0
    async_copy_b128(la0[0], gA0);
    async_copy_b128(la1[0], gA1);
    async_copy_b128(lb0[0], gB0);

    int cur = 0;
    for (int k0 = 0; k0 < Kd; k0 += BK) {
        wait_async0();          // my async writes into smem[cur] have landed
        __syncthreads();        // everyone's have, and prior reads are done

        if (k0 + BK < Kd) {     // prefetch next K-slab into the other buffer
            const int nxt = cur ^ 1;
            async_copy_b128(la0[nxt], gA0 + k0 + BK);
            async_copy_b128(la1[nxt], gA1 + k0 + BK);
            async_copy_b128(lb0[nxt], gB0 + k0 + BK);
        }

        const bf16* sA = &smem[cur][0];
        const bf16* sB = &smem[cur][A_ELEMS];
        v16bf a0 = load_frag(sA + (wm      + l15) * LDS_STRIDE + hlf * 8);
        v16bf a1 = load_frag(sA + (wm + 16 + l15) * LDS_STRIDE + hlf * 8);
        v16bf b0 = load_frag(sB + (wn      + l15) * LDS_STRIDE + hlf * 8);
        v16bf b1 = load_frag(sB + (wn + 16 + l15) * LDS_STRIDE + hlf * 8);

        acc[0][0] = __builtin_amdgcn_wmma_f32_16x16x32_bf16(false, a0, false, b0, (short)0, acc[0][0], false, false);
        acc[0][1] = __builtin_amdgcn_wmma_f32_16x16x32_bf16(false, a0, false, b1, (short)0, acc[0][1], false, false);
        acc[1][0] = __builtin_amdgcn_wmma_f32_16x16x32_bf16(false, a1, false, b0, (short)0, acc[1][0], false, false);
        acc[1][1] = __builtin_amdgcn_wmma_f32_16x16x32_bf16(false, a1, false, b1, (short)0, acc[1][1], false, false);

        cur ^= 1;
    }

    // C/D f32 16x16 layout: lane = N column, VGPR e = row (+8 for upper half)
    for (int i = 0; i < 2; ++i) {
        const int mb = bm + wm + i * 16 + hlf * 8;
        for (int j = 0; j < 2; ++j) {
            const int n = bn + wn + j * 16 + l15;
            epilogue_store<EPI>(Out, bias, N, mb, n, acc[i][j]);
        }
    }
}

// ---------------------------------------------------------------------------
// Direct-global WMMA GEMM (for small-N LoRA projections, N == 32).
// Each wave owns a 32x32 tile.
// ---------------------------------------------------------------------------
template<int EPI>
__global__ void __launch_bounds__(256)
gemm_bf16_wmma(const bf16* __restrict__ A, const bf16* __restrict__ Bt,
               void* __restrict__ Out, const float* __restrict__ bias,
               int M, int N, int Kd) {
    const int lane = threadIdx.x & 31;
    const int wave = threadIdx.x >> 5;
    const int l15  = lane & 15;
    const int hlf  = lane >> 4;

    const int tiles_m = M >> 5;
    const int tiles_n = N >> 5;
    const int tile = blockIdx.x * 8 + wave;
    if (tile >= tiles_m * tiles_n) return;
    const int tm = (tile % tiles_m) << 5;
    const int tn = (tile / tiles_m) << 5;

    v8f acc[2][2] = {{}, {}};

    const bf16* arow0 = A  + (size_t)(tm      + l15) * Kd + hlf * 8;
    const bf16* arow1 = A  + (size_t)(tm + 16 + l15) * Kd + hlf * 8;
    const bf16* bcol0 = Bt + (size_t)(tn      + l15) * Kd + hlf * 8;
    const bf16* bcol1 = Bt + (size_t)(tn + 16 + l15) * Kd + hlf * 8;

    for (int k0 = 0; k0 < Kd; k0 += 32) {
        v16bf a0 = load_frag(arow0 + k0);
        v16bf a1 = load_frag(arow1 + k0);
        v16bf b0 = load_frag(bcol0 + k0);
        v16bf b1 = load_frag(bcol1 + k0);
        if (k0 + 32 < Kd) {                       // -> global_prefetch_b8
            __builtin_prefetch(arow0 + k0 + 32, 0, 0);
            __builtin_prefetch(arow1 + k0 + 32, 0, 0);
        }
        acc[0][0] = __builtin_amdgcn_wmma_f32_16x16x32_bf16(false, a0, false, b0, (short)0, acc[0][0], false, false);
        acc[0][1] = __builtin_amdgcn_wmma_f32_16x16x32_bf16(false, a0, false, b1, (short)0, acc[0][1], false, false);
        acc[1][0] = __builtin_amdgcn_wmma_f32_16x16x32_bf16(false, a1, false, b0, (short)0, acc[1][0], false, false);
        acc[1][1] = __builtin_amdgcn_wmma_f32_16x16x32_bf16(false, a1, false, b1, (short)0, acc[1][1], false, false);
    }

    for (int i = 0; i < 2; ++i) {
        const int mb = tm + i * 16 + hlf * 8;
        for (int j = 0; j < 2; ++j) {
            const int n = tn + j * 16 + l15;
            epilogue_store<EPI>(Out, bias, N, mb, n, acc[i][j]);
        }
    }
}

// ---------------------------------------------------------------------------
// Weight transpose + f32 -> bf16:  Wt[n*rows + r] = W[r*cols + n]
// ---------------------------------------------------------------------------
__global__ void __launch_bounds__(256)
transpose_convert(const float* __restrict__ W, bf16* __restrict__ Wt,
                  int rows, int cols) {
    __shared__ float tile[32][33];
    const int tx = threadIdx.x & 31;
    const int ty = threadIdx.x >> 5;          // 0..7
    const int c0 = blockIdx.x * 32;
    const int r0 = blockIdx.y * 32;
    for (int j = ty; j < 32; j += 8) {
        const int r = r0 + j, c = c0 + tx;
        tile[j][tx] = (r < rows && c < cols) ? W[(size_t)r * cols + c] : 0.f;
    }
    __syncthreads();
    for (int j = ty; j < 32; j += 8) {
        const int n = c0 + j, r = r0 + tx;
        if (n < cols && r < rows) Wt[(size_t)n * rows + r] = (bf16)tile[tx][j];
    }
}

// ---------------------------------------------------------------------------
// Token-shift prep: delta = shift(x) - x ; five delta-mixed bf16 inputs
// ---------------------------------------------------------------------------
__global__ void __launch_bounds__(256)
prep_x(const float* __restrict__ x,
       const float* __restrict__ mu_r, const float* __restrict__ mu_k,
       const float* __restrict__ mu_v, const float* __restrict__ mu_g,
       const float* __restrict__ dd_mu,
       float* __restrict__ delta,
       bf16* __restrict__ xr, bf16* __restrict__ xk, bf16* __restrict__ xv,
       bf16* __restrict__ xg, bf16* __restrict__ xdd) {
    const size_t i = (size_t)blockIdx.x * blockDim.x + threadIdx.x;
    if (i >= (size_t)BT_ * D_) return;
    const int d   = (int)(i % D_);
    const size_t bt = i / D_;
    const int t   = (int)(bt % T_);
    const float xv_ = x[i];
    const float xp  = (t == 0) ? 0.f : x[i - D_];
    const float dl  = xp - xv_;
    delta[i] = dl;
    xr[i]  = (bf16)(xv_ + dl * mu_r[d]);
    xk[i]  = (bf16)(xv_ + dl * mu_k[d]);
    xv[i]  = (bf16)(xv_ + dl * mu_v[d]);
    xg[i]  = (bf16)(xv_ + dl * mu_g[d]);
    xdd[i] = (bf16)(xv_ + dl * dd_mu[d]);
}

// xw = x + delta * lora_dd   (bf16)
__global__ void __launch_bounds__(256)
mix_xw(const float* __restrict__ x, const float* __restrict__ delta,
       const float* __restrict__ lora, bf16* __restrict__ xw) {
    const size_t i = (size_t)blockIdx.x * blockDim.x + threadIdx.x;
    if (i >= (size_t)BT_ * D_) return;
    xw[i] = (bf16)(x[i] + delta[i] * lora[i]);
}

// ---------------------------------------------------------------------------
// Linear-attention recurrence. One block per (b,h); lane owns one DV column,
// 64-float state column lives in registers; r/k/decay broadcast via LDS.
// ---------------------------------------------------------------------------
__global__ void __launch_bounds__(128)
scan_kernel(const bf16* __restrict__ r, const bf16* __restrict__ k,
            const bf16* __restrict__ v, const float* __restrict__ decay,
            const float* __restrict__ bonus, float* __restrict__ o) {
    const int bh = blockIdx.x;
    const int b  = bh / H_;
    const int h  = bh % H_;
    const int tid = threadIdx.x;              // 0..127 -> DV column

    __shared__ float rs[DK_], ks[DK_], ds[DK_], us[DK_];
    if (tid < DK_) us[tid] = bonus[h * DK_ + tid];

    float hst[DK_];
    #pragma unroll
    for (int i = 0; i < DK_; ++i) hst[i] = 0.f;

    for (int t = 0; t < T_; ++t) {
        __syncthreads();
        if (tid < DK_) {
            const size_t idx = ((size_t)(b * T_ + t)) * K_ + (size_t)h * DK_ + tid;
            rs[tid] = (float)r[idx];
            ks[tid] = (float)k[idx];
            ds[tid] = decay[idx];
        }
        __syncthreads();
        const size_t vidx = ((size_t)(b * T_ + t)) * V_ + (size_t)h * DV_ + tid;
        const float vt = (float)v[vidx];
        float acc = 0.f;
        #pragma unroll
        for (int kk = 0; kk < DK_; ++kk) {
            const float kv = ks[kk] * vt;
            acc += rs[kk] * (hst[kk] + us[kk] * kv);   // read old state
            hst[kk] = hst[kk] * ds[kk] + kv;           // then decay+update
        }
        o[vidx] = acc;
    }
}

// ---------------------------------------------------------------------------
// RMS-norm over DV + SiLU gate
// ---------------------------------------------------------------------------
__global__ void __launch_bounds__(128)
norm_gate(const float* __restrict__ o, const bf16* __restrict__ g,
          const float* __restrict__ gw, bf16* __restrict__ out) {
    const size_t row = blockIdx.x;            // bt*H + h
    const int vcol = threadIdx.x;             // 0..127
    const size_t idx = row * DV_ + vcol;
    const float ov = o[idx];

    __shared__ float red[DV_];
    red[vcol] = ov * ov;
    __syncthreads();
    for (int s = 64; s > 0; s >>= 1) {
        if (vcol < s) red[vcol] += red[vcol + s];
        __syncthreads();
    }
    const float on = ov * rsqrtf(red[0] * (1.f / DV_) + EPS_) * gw[vcol];
    const float gg = (float)g[idx];
    const float gate = gg / (1.f + __expf(-gg));       // g * sigmoid(g)
    out[idx] = (bf16)(on * gate);
}

// ---------------------------------------------------------------------------
extern "C" void kernel_launch(void* const* d_in, const int* in_sizes, int n_in,
                              void* d_out, int out_size, void* d_ws, size_t ws_size,
                              hipStream_t stream) {
    const float* x       = (const float*)d_in[0];
    const float* W_r     = (const float*)d_in[1];
    const float* mu_r    = (const float*)d_in[2];
    const float* W_k     = (const float*)d_in[3];
    const float* mu_k    = (const float*)d_in[4];
    const float* W_v     = (const float*)d_in[5];
    const float* mu_v    = (const float*)d_in[6];
    const float* W_g     = (const float*)d_in[7];
    const float* mu_g    = (const float*)d_in[8];
    const float* dd_mu   = (const float*)d_in[9];
    const float* dd_W1   = (const float*)d_in[10];
    const float* dd_W2   = (const float*)d_in[11];
    const float* dd_lam  = (const float*)d_in[12];
    const float* w_W1    = (const float*)d_in[13];
    const float* w_W2    = (const float*)d_in[14];
    const float* w_lam   = (const float*)d_in[15];
    const float* bonus   = (const float*)d_in[16];
    const float* W_o     = (const float*)d_in[17];
    const float* g_norm  = (const float*)d_in[18];
    float* out = (float*)d_out;
    (void)in_sizes; (void)n_in; (void)out_size; (void)ws_size;

    char* ws = (char*)d_ws;
    size_t off = 0;
    auto alloc = [&](size_t bytes) -> void* {
        void* p = ws + off;
        off = (off + bytes + 255) & ~(size_t)255;
        return p;
    };

    bf16*  xr    = (bf16*) alloc((size_t)BT_ * D_ * 2);
    bf16*  xk    = (bf16*) alloc((size_t)BT_ * D_ * 2);
    bf16*  xv    = (bf16*) alloc((size_t)BT_ * D_ * 2);
    bf16*  xg    = (bf16*) alloc((size_t)BT_ * D_ * 2);
    bf16*  xdd   = (bf16*) alloc((size_t)BT_ * D_ * 2);
    float* delta = (float*)alloc((size_t)BT_ * D_ * 4);
    bf16*  WtR   = (bf16*) alloc((size_t)K_ * D_ * 2);
    bf16*  WtK   = (bf16*) alloc((size_t)K_ * D_ * 2);
    bf16*  WtV   = (bf16*) alloc((size_t)V_ * D_ * 2);
    bf16*  WtG   = (bf16*) alloc((size_t)V_ * D_ * 2);
    bf16*  Wtd1  = (bf16*) alloc((size_t)R_ * D_ * 2);
    bf16*  Wtd2  = (bf16*) alloc((size_t)D_ * R_ * 2);
    bf16*  Wtw1  = (bf16*) alloc((size_t)R_ * D_ * 2);
    bf16*  Wtw2  = (bf16*) alloc((size_t)K_ * R_ * 2);
    bf16*  WtO   = (bf16*) alloc((size_t)D_ * V_ * 2);
    bf16*  rb    = (bf16*) alloc((size_t)BT_ * K_ * 2);
    bf16*  kb    = (bf16*) alloc((size_t)BT_ * K_ * 2);
    bf16*  vb    = (bf16*) alloc((size_t)BT_ * V_ * 2);
    bf16*  gb    = (bf16*) alloc((size_t)BT_ * V_ * 2);
    float* decay = (float*)alloc((size_t)BT_ * K_ * 4);
    bf16*  t1    = (bf16*) alloc((size_t)BT_ * R_ * 2);
    float* lora  = (float*)alloc((size_t)BT_ * D_ * 4);
    bf16*  xw    = (bf16*) alloc((size_t)BT_ * D_ * 2);
    bf16*  t2    = (bf16*) alloc((size_t)BT_ * R_ * 2);
    float* ob    = (float*)alloc((size_t)BT_ * V_ * 4);
    bf16*  gated = (bf16*) alloc((size_t)BT_ * V_ * 2);

    // ---- weight transpose + bf16 conversion ----
    {
        dim3 blk(256);
        transpose_convert<<<dim3((K_+31)/32, (D_+31)/32), blk, 0, stream>>>(W_r,  WtR,  D_, K_);
        transpose_convert<<<dim3((K_+31)/32, (D_+31)/32), blk, 0, stream>>>(W_k,  WtK,  D_, K_);
        transpose_convert<<<dim3((V_+31)/32, (D_+31)/32), blk, 0, stream>>>(W_v,  WtV,  D_, V_);
        transpose_convert<<<dim3((V_+31)/32, (D_+31)/32), blk, 0, stream>>>(W_g,  WtG,  D_, V_);
        transpose_convert<<<dim3((R_+31)/32, (D_+31)/32), blk, 0, stream>>>(dd_W1, Wtd1, D_, R_);
        transpose_convert<<<dim3((D_+31)/32, (R_+31)/32), blk, 0, stream>>>(dd_W2, Wtd2, R_, D_);
        transpose_convert<<<dim3((R_+31)/32, (D_+31)/32), blk, 0, stream>>>(w_W1,  Wtw1, D_, R_);
        transpose_convert<<<dim3((K_+31)/32, (R_+31)/32), blk, 0, stream>>>(w_W2,  Wtw2, R_, K_);
        transpose_convert<<<dim3((D_+31)/32, (V_+31)/32), blk, 0, stream>>>(W_o,   WtO,  V_, D_);
    }

    // ---- token shift / delta mixes ----
    {
        const int n = BT_ * D_;
        prep_x<<<(n + 255) / 256, 256, 0, stream>>>(x, mu_r, mu_k, mu_v, mu_g, dd_mu,
                                                    delta, xr, xk, xv, xg, xdd);
    }

    // grid helpers
    auto nblk_lds   = [](int M, int N) { return (M / BM) * (N / BN); };
    auto nblk_small = [](int M, int N) { return ((M / 32) * (N / 32) + 7) / 8; };

    // ---- WMMA projections (async-LDS staged) ----
    gemm_bf16_wmma_lds<0><<<nblk_lds(BT_, K_), 256, 0, stream>>>(xr,  WtR,  rb,   nullptr, BT_, K_, D_);
    gemm_bf16_wmma_lds<0><<<nblk_lds(BT_, K_), 256, 0, stream>>>(xk,  WtK,  kb,   nullptr, BT_, K_, D_);
    gemm_bf16_wmma_lds<0><<<nblk_lds(BT_, V_), 256, 0, stream>>>(xv,  WtV,  vb,   nullptr, BT_, V_, D_);
    gemm_bf16_wmma_lds<0><<<nblk_lds(BT_, V_), 256, 0, stream>>>(xg,  WtG,  gb,   nullptr, BT_, V_, D_);

    // LoRA data-dependence chain (N=32 stages use the direct kernel)
    gemm_bf16_wmma<1><<<nblk_small(BT_, R_), 256, 0, stream>>>(xdd, Wtd1, t1, nullptr, BT_, R_, D_);
    gemm_bf16_wmma_lds<2><<<nblk_lds(BT_, D_), 256, 0, stream>>>(t1, Wtd2, lora, dd_lam, BT_, D_, R_);
    mix_xw<<<(BT_ * D_ + 255) / 256, 256, 0, stream>>>(x, delta, lora, xw);
    gemm_bf16_wmma<1><<<nblk_small(BT_, R_), 256, 0, stream>>>(xw, Wtw1, t2, nullptr, BT_, R_, D_);
    gemm_bf16_wmma_lds<3><<<nblk_lds(BT_, K_), 256, 0, stream>>>(t2, Wtw2, decay, w_lam, BT_, K_, R_);

    // ---- sequential recurrence: 64 independent (b,h) chains ----
    scan_kernel<<<B_ * H_, 128, 0, stream>>>(rb, kb, vb, decay, bonus, ob);

    // ---- RMS norm + SiLU gate ----
    norm_gate<<<BT_ * H_, 128, 0, stream>>>(ob, gb, g_norm, gated);

    // ---- output projection ----
    gemm_bf16_wmma_lds<4><<<nblk_lds(BT_, D_), 256, 0, stream>>>(gated, WtO, out, nullptr, BT_, D_, V_);
}
